// ResearchOptimalContrastiveLoss_34806414967388
// MI455X (gfx1250) — matrix operand order repaired and checked by reading, not compiled, not run on hardware
//
#include <hip/hip_runtime.h>
#include <hip/hip_bf16.h>

typedef __attribute__((ext_vector_type(16))) _Float16 v16h;
typedef __attribute__((ext_vector_type(8)))  _Float16 v8h;
typedef __attribute__((ext_vector_type(8)))  float    v8f;

static constexpr int kD  = 512;    // feature dim (compile-time K for the GEMM)
static constexpr int kB  = 1024;   // batch
static constexpr int kM  = 16384;  // buffer rows
static constexpr int kC  = 100;    // classes
static constexpr int kL  = 50;     // learned classes
static constexpr int kNP = 64;     // padded proto rows (50 -> 64)
static constexpr float kTempInv = 5.0f;   // 1/0.2
static constexpr float kClip    = 10.0f;

// Monotonic float->uint transform (descending order == descending unsigned)
__device__ __forceinline__ unsigned fxform(float s) {
  unsigned b = __float_as_uint(s);
  return b ^ ((unsigned)((int)b >> 31) | 0x80000000u);
}
__device__ __forceinline__ float funxform(unsigned u) {
  unsigned b = (u & 0x80000000u) ? (u ^ 0x80000000u) : ~u;
  return __uint_as_float(b);
}

// Assemble a 16-half WMMA fragment from two contiguous 8-half (16B) chunks.
// hioff = 16 for both A and B fragments (K-interleave {0-7}/{16-23}).
__device__ __forceinline__ v16h load_frag(const _Float16* __restrict__ p, int hioff) {
  v8h lo = *(const v8h*)(p);
  v8h hi = *(const v8h*)(p + hioff);
  v16h r;
#pragma unroll
  for (int i = 0; i < 8; ++i) { r[i] = lo[i]; r[i + 8] = hi[i]; }
  return r;
}

// ---------------------------------------------------------------------------
// Row L2-normalize f32 -> f16, optional gather via idx; rows >= nvalid zeroed.
// One wave per row (8 waves / 256-thread block).
// ---------------------------------------------------------------------------
__global__ void __launch_bounds__(256)
norm_rows_f16(const float* __restrict__ src, _Float16* __restrict__ dst,
              const int* __restrict__ idx, int nrows, int nvalid) {
  const int lane = threadIdx.x & 31;
  const int wave = threadIdx.x >> 5;
  const int row  = blockIdx.x * 8 + wave;
  if (row >= nrows) return;
  int srow = row;
  bool valid = true;
  if (idx) { if (row < nvalid) srow = idx[row]; else valid = false; }
  float vals[16];
  float ss = 0.f;
  if (valid) {
    const float* p = src + (size_t)srow * kD;
#pragma unroll
    for (int i = 0; i < 16; ++i) { float x = p[lane + i * 32]; vals[i] = x; ss += x * x; }
  } else {
#pragma unroll
    for (int i = 0; i < 16; ++i) vals[i] = 0.f;
  }
#pragma unroll
  for (int o = 16; o > 0; o >>= 1) ss += __shfl_xor(ss, o, 32);
  const float scale = valid ? 1.0f / fmaxf(sqrtf(ss), 1e-12f) : 0.f;
  _Float16* q = dst + (size_t)row * kD;
#pragma unroll
  for (int i = 0; i < 16; ++i) q[lane + i * 32] = (_Float16)(vals[i] * scale);
}

// ---------------------------------------------------------------------------
// C[M x N] = clip( (A[M x kD] * B[N x kD]^T) * kTempInv, +-kClip ) via WMMA f16.
// Block = 128 threads (4 waves); wave w does rows [by*64 + 16w, +16), cols
// [bx*64, +64) as 4 accumulators. K = kD is compile-time: the k-loop fully
// unrolls into a flat branch-free block so the scheduler pipelines loads
// across WMMA bundles (partial s_wait_loadcnt) without register copies.
// ---------------------------------------------------------------------------
__global__ void __launch_bounds__(128)
wmma_gemm_nt(const _Float16* __restrict__ A, const _Float16* __restrict__ Bm,
             float* __restrict__ Cout, int Ncols) {
  const int lane = threadIdx.x & 31;
  const int wave = threadIdx.x >> 5;
  const int r0   = blockIdx.y * 64 + wave * 16;
  const int c0   = blockIdx.x * 64;
  const int lrow = lane & 15;
  const int ah   = (lane >> 4) * 8;   // A K-half offset
  const int bh   = (lane >> 4) * 8;   // B K-half offset
  v8f acc[4] = {};
  const _Float16* arow = A + (size_t)(r0 + lrow) * kD + ah;
  const _Float16* brow[4];
#pragma unroll
  for (int t = 0; t < 4; ++t) brow[t] = Bm + (size_t)(c0 + 16 * t + lrow) * kD + bh;

#pragma unroll
  for (int k0 = 0; k0 < kD; k0 += 32) {
    v16h a = load_frag(arow + k0, 16);
#pragma unroll
    for (int t = 0; t < 4; ++t) {
      v16h b = load_frag(brow[t] + k0, 16);
      acc[t] = __builtin_amdgcn_wmma_f32_16x16x32_f16(false, a, false, b,
                                                      (short)0, acc[t], false, false);
    }
  }

  const int mb = (lane >> 4) * 8;
#pragma unroll
  for (int t = 0; t < 4; ++t) {
#pragma unroll
    for (int v = 0; v < 8; ++v) {
      float s = fminf(fmaxf(acc[t][v] * kTempInv, -kClip), kClip);
      Cout[(size_t)(r0 + mb + v) * Ncols + (c0 + 16 * t + lrow)] = s;
    }
  }
}

// ---------------------------------------------------------------------------
// scores[i] = sigmoid(features[i] . W + b); one wave per sample.
// ---------------------------------------------------------------------------
__global__ void __launch_bounds__(256)
scores_kernel(const float* __restrict__ f, const float* __restrict__ W,
              const float* __restrict__ b, float* __restrict__ scores) {
  const int lane = threadIdx.x & 31;
  const int wave = threadIdx.x >> 5;
  const int i = blockIdx.x * 8 + wave;
  if (i >= kB) return;
  const float* fr = f + (size_t)i * kD;
  float dot = 0.f;
#pragma unroll
  for (int t = 0; t < 16; ++t) { int d = lane + t * 32; dot += fr[d] * W[d]; }
#pragma unroll
  for (int o = 16; o > 0; o >>= 1) dot += __shfl_xor(dot, o, 32);
  if (lane == 0) scores[i] = 1.f / (1.f + expf(-(dot + b[0])));
}

// ---------------------------------------------------------------------------
// Hard-negative loss per row: exact k-th-largest via 4-pass byte radix select
// over monotonic float bits, then neg_exp = sum_{v>T} exp(v) + ties*exp(T).
// One 256-thread block per row; sims row is L2-resident.
// acc: [0]=hard_sum [1]=hard_cnt
// ---------------------------------------------------------------------------
__global__ void __launch_bounds__(256)
hard_neg_kernel(const float* __restrict__ sims, const int* __restrict__ labels,
                const int* __restrict__ blabels, float* __restrict__ acc) {
  __shared__ unsigned hist[256];
  __shared__ float    sredf[8];
  __shared__ unsigned sredu[16];
  __shared__ float    s_pos_sum;
  __shared__ unsigned s_npos;
  __shared__ int      s_k;
  __shared__ unsigned s_prefix;
  __shared__ unsigned s_krem;

  const int row = blockIdx.x;
  const int tid = threadIdx.x;
  const int lab = labels[row];
  const float* srow = sims + (size_t)row * kM;

  for (int i = tid; i < 256; i += 256) hist[i] = 0u;
  __syncthreads();

  // Pass 0: pos stats + top-byte histogram of negatives
  float pos_sum = 0.f;
  unsigned npos = 0, nneg = 0;
  for (int j = tid; j < kM; j += 256) {
    if (j + 4096 < kM) __builtin_prefetch(srow + j + 4096, 0, 0);  // global_prefetch_b8
    float s = srow[j];
    if (blabels[j] == lab) { pos_sum += expf(s); ++npos; }
    else { ++nneg; atomicAdd(&hist[fxform(s) >> 24], 1u); }
  }
  {
    float v = pos_sum; unsigned un = npos, ug = nneg;
#pragma unroll
    for (int o = 16; o > 0; o >>= 1) {
      v += __shfl_xor(v, o, 32);
      un += __shfl_xor(un, o, 32);
      ug += __shfl_xor(ug, o, 32);
    }
    if ((tid & 31) == 0) { sredf[tid >> 5] = v; sredu[tid >> 5] = un; sredu[8 + (tid >> 5)] = ug; }
  }
  __syncthreads();
  if (tid == 0) {
    float ps = 0.f; unsigned np = 0, ng = 0;
    for (int i = 0; i < 8; ++i) { ps += sredf[i]; np += sredu[i]; ng += sredu[8 + i]; }
    s_pos_sum = ps; s_npos = np;
    int k = (int)(0.7f * (float)ng);   // RATIO * n_neg, truncated
    s_k = k;
    if (k > 0) {
      unsigned cum = 0, target = (unsigned)k, sel = 0, rem = 0;
      for (int bb = 255; bb >= 0; --bb) {
        unsigned nb = hist[bb];
        if (cum + nb >= target) { sel = (unsigned)bb; rem = target - cum; break; }
        cum += nb;
      }
      s_prefix = sel << 24;
      s_krem   = rem;
    }
  }
  __syncthreads();

  const int k = s_k;
  // Radix refinement passes (bytes 2,1,0)
  for (int p = 1; p <= 3 && k > 0; ++p) {
    __syncthreads();
    for (int i = tid; i < 256; i += 256) hist[i] = 0u;
    __syncthreads();
    const unsigned prefix = s_prefix;
    const int topShift  = 32 - 8 * p;
    const int byteShift = 24 - 8 * p;
    for (int j = tid; j < kM; j += 256) {
      if (blabels[j] != lab) {
        unsigned u = fxform(srow[j]);
        if (((u ^ prefix) >> topShift) == 0u)
          atomicAdd(&hist[(u >> byteShift) & 255u], 1u);
      }
    }
    __syncthreads();
    if (tid == 0) {
      unsigned cum = 0, target = s_krem, sel = 0, rem = target;
      for (int bb = 255; bb >= 0; --bb) {
        unsigned nb = hist[bb];
        if (cum + nb >= target) { sel = (unsigned)bb; rem = target - cum; break; }
        cum += nb;
      }
      s_prefix = prefix | (sel << byteShift);
      s_krem   = rem;
    }
    __syncthreads();
  }

  // Final pass: sum exp over strictly-greater values
  if (k > 0) {
    const unsigned T = s_prefix;
    float gt = 0.f;
    for (int j = tid; j < kM; j += 256) {
      if (blabels[j] != lab) {
        float s = srow[j];
        if (fxform(s) > T) gt += expf(s);
      }
    }
#pragma unroll
    for (int o = 16; o > 0; o >>= 1) gt += __shfl_xor(gt, o, 32);
    __syncthreads();
    if ((tid & 31) == 0) sredf[tid >> 5] = gt;
    __syncthreads();
  }
  if (tid == 0) {
    float ne = 0.f;
    if (k > 0) {
      float g = 0.f;
      for (int i = 0; i < 8; ++i) g += sredf[i];
      ne = g + (float)s_krem * expf(funxform(s_prefix));
    }
    float npos_f  = fmaxf((float)s_npos, 1.f);
    float pos_exp = s_pos_sum / npos_f;
    bool valid = (k > 0) && (s_npos > 0);
    float denom = fmaxf(pos_exp + ne, 1e-8f);
    float ratio = fminf(fmaxf(pos_exp / denom, 1e-8f), 1.f);
    float loss  = fminf(fmaxf(-logf(ratio), 0.f), 5.f);
    if (valid) { atomicAdd(&acc[0], loss); atomicAdd(&acc[1], 1.f); }
  }
}

// ---------------------------------------------------------------------------
// PRD KL per row over 50 valid columns (padded to 64); one wave per row.
// acc[2] += row KL
// ---------------------------------------------------------------------------
__global__ void __launch_bounds__(256)
prd_kl_kernel(const float* __restrict__ olds, const float* __restrict__ news,
              float* __restrict__ acc) {
  const int lane = threadIdx.x & 31;
  const int wave = threadIdx.x >> 5;
  const int row = blockIdx.x * 8 + wave;
  if (row >= kB) return;
  const float* orow = olds + (size_t)row * kNP;
  const float* nrow = news + (size_t)row * kNP;
  const bool v0 = lane < kL;
  const bool v1 = (lane + 32) < kL;
  float o0 = v0 ? orow[lane] : -1e30f;
  float o1 = v1 ? orow[lane + 32] : -1e30f;
  float n0 = v0 ? nrow[lane] : -1e30f;
  float n1 = v1 ? nrow[lane + 32] : -1e30f;
  float mo = fmaxf(o0, o1), mn = fmaxf(n0, n1);
#pragma unroll
  for (int o = 16; o > 0; o >>= 1) {
    mo = fmaxf(mo, __shfl_xor(mo, o, 32));
    mn = fmaxf(mn, __shfl_xor(mn, o, 32));
  }
  float eo = (v0 ? expf(o0 - mo) : 0.f) + (v1 ? expf(o1 - mo) : 0.f);
  float en = (v0 ? expf(n0 - mn) : 0.f) + (v1 ? expf(n1 - mn) : 0.f);
#pragma unroll
  for (int o = 16; o > 0; o >>= 1) { eo += __shfl_xor(eo, o, 32); en += __shfl_xor(en, o, 32); }
  float lzo = mo + logf(eo);
  float lzn = mn + logf(en);
  float kl = 0.f;
  if (v0) { float lo = o0 - lzo, ln = n0 - lzn; kl += expf(lo) * (lo - ln); }
  if (v1) { float lo = o1 - lzo, ln = n1 - lzn; kl += expf(lo) * (lo - ln); }
#pragma unroll
  for (int o = 16; o > 0; o >>= 1) kl += __shfl_xor(kl, o, 32);
  if (lane == 0) atomicAdd(&acc[2], kl);
}

// ---------------------------------------------------------------------------
// Boundary loss: one 128-thread block per class.
// acc[3]=bl_sum, acc[4]=valid_class_count
// ---------------------------------------------------------------------------
__global__ void __launch_bounds__(128)
boundary_kernel(const float* __restrict__ f, const int* __restrict__ labels,
                const float* __restrict__ scores, float* __restrict__ acc) {
  __shared__ float sumf[kD];
  __shared__ float red[128];
  __shared__ unsigned s_n;
  __shared__ float s_ssum;
  const int c = blockIdx.x;
  const int tid = threadIdx.x;
  for (int d = tid; d < kD; d += 128) sumf[d] = 0.f;
  if (tid == 0) { s_n = 0u; s_ssum = 0.f; }
  __syncthreads();
  float ssq = 0.f;
  for (int i = 0; i < kB; ++i) {
    if (labels[i] == c) {               // block-uniform branch
      const float* fr = f + (size_t)i * kD;
      for (int d = tid; d < kD; d += 128) { float x = fr[d]; sumf[d] += x; ssq += x * x; }
      if (tid == 0) { s_n++; s_ssum += scores[i]; }
    }
  }
  __syncthreads();
  float sq2 = 0.f;
  for (int d = tid; d < kD; d += 128) sq2 += sumf[d] * sumf[d];
  red[tid] = ssq;
  __syncthreads();
  for (int off = 64; off > 0; off >>= 1) { if (tid < off) red[tid] += red[tid + off]; __syncthreads(); }
  const float ssq_tot = red[0];
  __syncthreads();
  red[tid] = sq2;
  __syncthreads();
  for (int off = 64; off > 0; off >>= 1) { if (tid < off) red[tid] += red[tid + off]; __syncthreads(); }
  const float sq2_tot = red[0];
  if (tid == 0) {
    float n = (float)s_n;
    float safe_n = fmaxf(n, 1.f);
    float nm1 = fmaxf(n - 1.f, 1.f);
    float varmean = (ssq_tot - n * sq2_tot / (safe_n * safe_n)) / nm1 / (float)kD;
    varmean = fminf(fmaxf(varmean, 1e-6f), 100.f);
    float target = 1.f / (1.f + expf(-varmean));
    float mean_s = s_ssum / safe_n;
    float dd = mean_s - target;
    float bl = fminf(fmaxf(dd * dd, 0.f), 2.f);
    if (s_n > 1u) { atomicAdd(&acc[3], bl); atomicAdd(&acc[4], 1.f); }
  }
}

__global__ void init_acc_kernel(float* __restrict__ acc) {
  if (threadIdx.x < 8) acc[threadIdx.x] = 0.f;
}

__global__ void finalize_kernel(const float* __restrict__ acc, float* __restrict__ out) {
  float hard = acc[0] / fmaxf(acc[1], 1.f);
  float kl = acc[2] / (float)kB;
  kl = fminf(fmaxf(kl, 0.f), 5.f);
  float boundary = acc[3] / fmaxf(acc[4], 1.f);
  out[0] = hard + 0.1f * boundary + 0.2f * kl;
}

extern "C" void kernel_launch(void* const* d_in, const int* in_sizes, int n_in,
                              void* d_out, int out_size, void* d_ws, size_t ws_size,
                              hipStream_t stream) {
  (void)in_sizes; (void)n_in; (void)out_size; (void)ws_size;
  const float* features = (const float*)d_in[0];
  const float* buf_feat = (const float*)d_in[1];
  const float* protos   = (const float*)d_in[2];
  const float* old_prot = (const float*)d_in[3];
  const float* W        = (const float*)d_in[4];
  const float* b        = (const float*)d_in[5];
  const int*   labels   = (const int*)d_in[6];
  const int*   blabels  = (const int*)d_in[7];
  const int*   learned  = (const int*)d_in[8];
  float* out = (float*)d_out;

  char* ws = (char*)d_ws;
  size_t off = 0;
  float*    sims = (float*)(ws + off);    off += (size_t)kB * kM * 4;     // 64 MB
  _Float16* fn   = (_Float16*)(ws + off); off += (size_t)kB * kD * 2;     // 1 MB
  _Float16* bfn  = (_Float16*)(ws + off); off += (size_t)kM * kD * 2;     // 16 MB
  _Float16* opn  = (_Float16*)(ws + off); off += (size_t)kNP * kD * 2;
  _Float16* npn  = (_Float16*)(ws + off); off += (size_t)kNP * kD * 2;
  float*    olds = (float*)(ws + off);    off += (size_t)kB * kNP * 4;
  float*    news = (float*)(ws + off);    off += (size_t)kB * kNP * 4;
  float*    scr  = (float*)(ws + off);    off += (size_t)kB * 4;
  float*    acc  = (float*)(ws + off);    off += 64;

  init_acc_kernel<<<1, 32, 0, stream>>>(acc);
  norm_rows_f16<<<kB / 8, 256, 0, stream>>>(features, fn, nullptr, kB, kB);
  norm_rows_f16<<<kM / 8, 256, 0, stream>>>(buf_feat, bfn, nullptr, kM, kM);
  norm_rows_f16<<<kNP / 8, 256, 0, stream>>>(old_prot, opn, learned, kNP, kL);
  norm_rows_f16<<<kNP / 8, 256, 0, stream>>>(protos,   npn, learned, kNP, kL);
  wmma_gemm_nt<<<dim3(kM / 64, kB / 64), 128, 0, stream>>>(fn, bfn, sims, kM);
  wmma_gemm_nt<<<dim3(1, kB / 64), 128, 0, stream>>>(fn, opn, olds, kNP);
  wmma_gemm_nt<<<dim3(1, kB / 64), 128, 0, stream>>>(fn, npn, news, kNP);
  scores_kernel<<<kB / 8, 256, 0, stream>>>(features, W, b, scr);
  hard_neg_kernel<<<kB, 256, 0, stream>>>(sims, labels, blabels, acc);
  prd_kl_kernel<<<kB / 8, 256, 0, stream>>>(olds, news, acc);
  boundary_kernel<<<kC, 128, 0, stream>>>(features, labels, scr, acc);
  finalize_kernel<<<1, 1, 0, stream>>>(acc, out);
}
